// NormalizedCutLoss_31868657336630
// MI455X (gfx1250) — compile-verified, matrix operand order checked by reading, not run on hardware
//
#include <hip/hip_runtime.h>
#include <hip/hip_bf16.h>

typedef __attribute__((ext_vector_type(16))) _Float16 v16h;
typedef __attribute__((ext_vector_type(8)))  _Float16 v8h;
typedef __attribute__((ext_vector_type(8)))  float    v8f;
typedef __attribute__((ext_vector_type(2)))  float    v2f;

// Problem constants (from reference): 8 images, 21 seg channels, 128->64 resize.
#define NIMG 8
#define KSEG 21
#define PPX  4096            // 64*64 pixels per image
#define KPADF 8              // feature K padded 5 -> 8 (two K=4 f32 WMMAs)
#define KPADS 32             // seg K padded 21 -> 32 (one K=32 f16 WMMA)
#define TILE 128             // pair-matrix tile edge per block
#define SSTR 40              // LDS row stride (halfs) for S tiles: 80B, 16B-aligned
#define FSTR 10              // LDS row stride (floats) for F tiles: 40B, 8B-aligned

// raw hardware exp2 (v_exp_f32) -- our arg is <= 0, underflow->0 is fine
__device__ __forceinline__ float hw_exp2(float x) {
#if __has_builtin(__builtin_amdgcn_exp2f)
  return __builtin_amdgcn_exp2f(x);
#else
  return exp2f(x);
#endif
}

// ---------------------------------------------------------------------------
// Prep: padded f32 features, padded f16 seg, per-pixel ||f||^2, seg colsums.
// Exact bilinear x0.5 == 2x2 average; nearest == stride-2 sample.
// ---------------------------------------------------------------------------
__global__ __launch_bounds__(256) void nc_prep(
    const float* __restrict__ img,   // (8,3,128,128)
    const float* __restrict__ seg,   // (8,21,128,128)
    const float* __restrict__ roi,   // (8,128,128)
    float* __restrict__ Ff,          // (8, KPADF, 4096)
    _Float16* __restrict__ Sh,       // (8, KPADS, 4096)
    float* __restrict__ sq,          // (8, 4096)
    float* __restrict__ colS)        // (8, 4096)
{
  int idx = blockIdx.x * 256 + threadIdx.x;     // 0 .. 8*4096-1
  int n = idx >> 12;
  int p = idx & (PPX - 1);
  int y = p >> 6, x = p & 63;
  int src = (2 * y) * 128 + 2 * x;

  const float inv_rgb = 1.0f / 15.0f;           // 1/SIGMA_RGB
  const float inv_xy  = 1.0f / 40.0f;           // 1/(SIGMA_XY*SCALE)

  float f0 = img[(size_t)n * 3 * 16384 + 0 * 16384 + src] * inv_rgb;
  float f1 = img[(size_t)n * 3 * 16384 + 1 * 16384 + src] * inv_rgb;
  float f2 = img[(size_t)n * 3 * 16384 + 2 * 16384 + src] * inv_rgb;
  float f3 = (float)x * inv_xy;
  float f4 = (float)y * inv_xy;

  float* Fn = Ff + (size_t)n * KPADF * PPX;
  Fn[0 * PPX + p] = f0;  Fn[1 * PPX + p] = f1;  Fn[2 * PPX + p] = f2;
  Fn[3 * PPX + p] = f3;  Fn[4 * PPX + p] = f4;
  Fn[5 * PPX + p] = 0.f; Fn[6 * PPX + p] = 0.f; Fn[7 * PPX + p] = 0.f;

  sq[n * PPX + p] = f0 * f0 + f1 * f1 + f2 * f2 + f3 * f3 + f4 * f4;

  float rv = roi[(size_t)n * 16384 + src];
  _Float16* Sn = Sh + (size_t)n * KPADS * PPX;
  float cs = 0.f;
#pragma unroll 4
  for (int k = 0; k < KPADS; ++k) {
    float v = 0.f;
    if (k < KSEG) {
      const float* sp = seg + ((size_t)n * KSEG + k) * 16384 + src;
      v = 0.25f * (sp[0] + sp[1] + sp[128] + sp[129]) * rv;  // bilinear frac=0.5
      cs += v;
    }
    Sn[k * PPX + p] = (_Float16)v;
  }
  colS[n * PPX + p] = cs;
}

// ---------------------------------------------------------------------------
// Main: one block per 128x128 tile of the P x P pair matrix of one image.
// LDS tiles stored pixel-major / K-contiguous so WMMA fragments are loaded
// with 2x ds_load_b128 (S, f16) and 2x ds_load_b64 (F, f32) per fragment.
// ---------------------------------------------------------------------------
__global__ __launch_bounds__(256) void nc_main(
    const float* __restrict__ Ff, const _Float16* __restrict__ Sh,
    const float* __restrict__ sq, const float* __restrict__ colS,
    float* __restrict__ partials)
{
  __shared__ _Float16 sSI[TILE][SSTR];   // [pixel][k], k runs contiguous
  __shared__ _Float16 sSJ[TILE][SSTR];
  __shared__ float    sFI[TILE][FSTR];
  __shared__ float    sFJ[TILE][FSTR];
  __shared__ float    sSqI[TILE], sSqJ[TILE], sColI[TILE];
  __shared__ float    sRed[2 * 256];

  const int bid  = blockIdx.x;
  const int n    = bid >> 10;            // 1024 tiles per image (32x32)
  const int tile = bid & 1023;
  const int i0   = (tile >> 5) * TILE;
  const int j0   = (tile & 31) * TILE;
  const int t    = threadIdx.x;

  // ---- cooperative LDS fill with on-the-fly transpose ----
  {
    const uint32_t* gSI = (const uint32_t*)(Sh + (size_t)n * KPADS * PPX);
    unsigned short* uSI = (unsigned short*)&sSI[0][0];
    unsigned short* uSJ = (unsigned short*)&sSJ[0][0];
#pragma unroll
    for (int u = 0; u < 8; ++u) {                 // 2048 dwords per matrix
      int f = u * 256 + t;
      int k = f >> 6, pair = f & 63;              // dword = pixels 2*pair,2*pair+1
      uint32_t di = gSI[k * 2048 + (i0 >> 1) + pair];
      uint32_t dj = gSI[k * 2048 + (j0 >> 1) + pair];
      int m0 = 2 * pair;
      uSI[(m0    ) * SSTR + k] = (unsigned short)(di);
      uSI[(m0 + 1) * SSTR + k] = (unsigned short)(di >> 16);
      uSJ[(m0    ) * SSTR + k] = (unsigned short)(dj);
      uSJ[(m0 + 1) * SSTR + k] = (unsigned short)(dj >> 16);
    }
    const float* gF = Ff + (size_t)n * KPADF * PPX;
#pragma unroll
    for (int u = 0; u < 4; ++u) {                 // 1024 floats per matrix
      int f = u * 256 + t;
      int k = f >> 7, m = f & 127;
      sFI[m][k] = gF[k * PPX + i0 + m];
      sFJ[m][k] = gF[k * PPX + j0 + m];
    }
    if (t < TILE) {
      sSqI[t]  = sq[n * PPX + i0 + t];
      sSqJ[t]  = sq[n * PPX + j0 + t];
      sColI[t] = colS[n * PPX + i0 + t];
    }
  }
  __syncthreads();

  const int w    = t >> 5;           // wave id: rows [16w, 16w+16)
  const int lane = t & 31;
  const int g    = lane >> 4;        // CDNA5 wave32 WMMA lane group
  const int ln   = lane & 15;
  const int mrow = w * 16 + ln;      // A-fragment row held by this lane

  // A fragment (SᵀS): lane(g,ln) K-runs [8g,8g+8) and [16+8g,16+8g+8)
  v8h a0 = *(const v8h*)&sSI[mrow][8 * g];
  v8h a1 = *(const v8h*)&sSI[mrow][16 + 8 * g];
  v16h aS = __builtin_shufflevector(a0, a1,
      0, 1, 2, 3, 4, 5, 6, 7, 8, 9, 10, 11, 12, 13, 14, 15);
#if __has_builtin(__builtin_amdgcn_wmma_f32_16x16x4_f32)
  // A fragments (fᵀf): K-runs {2g,2g+1} and {4+2g,5+2g}
  v2f aF0 = *(const v2f*)&sFI[mrow][2 * g];
  v2f aF1 = *(const v2f*)&sFI[mrow][4 + 2 * g];
#endif
  // D-element row within this wave's strip: m = r + 8g
  float sqi[8], colv[8];
#pragma unroll
  for (int r = 0; r < 8; ++r) {
    sqi[r]  = sSqI[w * 16 + 8 * g + r];
    colv[r] = sColI[w * 16 + 8 * g + r];
  }

  float numA = 0.f, denA = 0.f;
  const v8f zeroC = {0.f, 0.f, 0.f, 0.f, 0.f, 0.f, 0.f, 0.f};
  const float cexp = -0.72134752044448170f;   // -0.5 * log2(e); v_exp_f32 is exp2

#pragma unroll
  for (int jt = 0; jt < 8; ++jt) {            // full unroll: LDS offsets become imms
    const int ncol = jt * 16 + ln;            // B-fragment column
    v8h b0 = *(const v8h*)&sSJ[ncol][8 * g];
    v8h b1 = *(const v8h*)&sSJ[ncol][16 + 8 * g];
    v16h bS = __builtin_shufflevector(b0, b1,
        0, 1, 2, 3, 4, 5, 6, 7, 8, 9, 10, 11, 12, 13, 14, 15);

    // seg Gram: G = SᵀS, one V_WMMA_F32_16X16X32_F16
    v8f gS = __builtin_amdgcn_wmma_f32_16x16x32_f16(
        false, aS, false, bS, (short)0, zeroC, false, false);

    v8f gF;
#if __has_builtin(__builtin_amdgcn_wmma_f32_16x16x4_f32)
    v2f bF0 = *(const v2f*)&sFJ[ncol][2 * g];
    v2f bF1 = *(const v2f*)&sFJ[ncol][4 + 2 * g];
    gF = __builtin_amdgcn_wmma_f32_16x16x4_f32(
        false, aF0, false, bF0, (short)0, zeroC, false, false);
    gF = __builtin_amdgcn_wmma_f32_16x16x4_f32(
        false, aF1, false, bF1, (short)0, gF, false, false);
#else
#pragma unroll
    for (int r = 0; r < 8; ++r) {
      float acc = 0.f;
#pragma unroll
      for (int k = 0; k < 5; ++k)
        acc = fmaf(sFI[w * 16 + 8 * g + r][k], sFJ[ncol][k], acc);
      gF[r] = acc;
    }
#endif
    const float sqj = sSqJ[ncol];
#pragma unroll
    for (int r = 0; r < 8; ++r) {
      float d2 = sqi[r] + sqj - 2.0f * gF[r];
      d2 = fmaxf(d2, 0.0f);
      float wv = hw_exp2(cexp * d2);          // mul + raw v_exp_f32 (no libm fixup)
      numA = fmaf(wv, gS[r],   numA);
      denA = fmaf(wv, colv[r], denA);
    }
  }

  // deterministic block reduction -> per-block partials (no float atomics)
  sRed[t] = numA;
  sRed[256 + t] = denA;
  __syncthreads();
#pragma unroll
  for (int s = 128; s > 0; s >>= 1) {
    if (t < s) {
      sRed[t]       += sRed[t + s];
      sRed[256 + t] += sRed[256 + t + s];
    }
    __syncthreads();
  }
  if (t == 0) {
    partials[2 * bid]     = sRed[0];
    partials[2 * bid + 1] = sRed[256];
  }
}

// ---------------------------------------------------------------------------
// Final: fixed-order reduction of all block partials -> loss scalar
// ---------------------------------------------------------------------------
__global__ __launch_bounds__(256) void nc_final(
    const float* __restrict__ partials, float* __restrict__ out, int nb)
{
  __shared__ float rn[256], rd[256];
  int t = threadIdx.x;
  float sn = 0.f, sd = 0.f;
  for (int i = t; i < nb; i += 256) {
    sn += partials[2 * i];
    sd += partials[2 * i + 1];
  }
  rn[t] = sn; rd[t] = sd;
  __syncthreads();
#pragma unroll
  for (int s = 128; s > 0; s >>= 1) {
    if (t < s) { rn[t] += rn[t + s]; rd[t] += rd[t + s]; }
    __syncthreads();
  }
  if (t == 0) out[0] = -(rn[0] / rd[0]) * (1.0f / (float)NIMG);  // WEIGHT=1
}

// ---------------------------------------------------------------------------
extern "C" void kernel_launch(void* const* d_in, const int* in_sizes, int n_in,
                              void* d_out, int out_size, void* d_ws, size_t ws_size,
                              hipStream_t stream) {
  const float* images = (const float*)d_in[0];   // (8,3,128,128) f32
  const float* segs   = (const float*)d_in[1];   // (8,21,128,128) f32
  const float* rois   = (const float*)d_in[2];   // (8,128,128) f32
  float* out = (float*)d_out;

  // workspace layout (bytes)
  char* base = (char*)d_ws;
  float*    Ff       = (float*)(base);                    // 8*8*4096*4  = 1,048,576
  _Float16* Sh       = (_Float16*)(base + 1048576);       // 8*32*4096*2 = 2,097,152
  float*    sq       = (float*)(base + 3145728);          // 8*4096*4    =   131,072
  float*    colS     = (float*)(base + 3276800);          // 8*4096*4    =   131,072
  float*    partials = (float*)(base + 3407872);          // 8192*2*4    =    65,536

  nc_prep<<<(NIMG * PPX) / 256, 256, 0, stream>>>(images, segs, rois,
                                                  Ff, Sh, sq, colS);

  const int nblocks = NIMG * (PPX / TILE) * (PPX / TILE); // 8 * 32 * 32 = 8192
  nc_main<<<nblocks, 256, 0, stream>>>(Ff, Sh, sq, colS, partials);

  nc_final<<<1, 256, 0, stream>>>(partials, out, nblocks);
}
